// FFTTransformerBlock_12902081757736
// MI455X (gfx1250) — compile-verified, matrix operand order checked by reading, not run on hardware
//
#include <hip/hip_runtime.h>
#include <hip/hip_bf16.h>
#include <math.h>

// ---------------- types ----------------
typedef _Float16 h16;
typedef __attribute__((ext_vector_type(2)))  _Float16 v2h;
typedef __attribute__((ext_vector_type(16))) _Float16 v16h;
typedef __attribute__((ext_vector_type(8)))  float    v8f;
typedef __attribute__((ext_vector_type(4)))  unsigned int u32x4;
typedef __attribute__((ext_vector_type(8)))  unsigned int u32x8;

#define HWSZ  (256 * 256)
#define NB    2
#define NPOS  (NB * HWSZ)

#define MODE_LN        0
#define MODE_LN_PROD   1
#define MODE_GELU_PROD 2

// =====================================================================
// Fused (TDM tile stage -> transform -> f16 LDS -> WMMA GEMM) kernel
//   out[b,o,hw] = resid[b,o,hw] + bias[o] + sum_c T(in)[b,c,hw] * W[o,c]
//   T = LayerNorm(ch) | LayerNorm(ch)*in1 | gelu(in0)*in1
// 256 threads = 8 wave32.  CIN*PT == 8192, O%16 == 0.
// =====================================================================
template <int MODE, int CIN, int O, int PT, bool HASR>
__global__ __launch_bounds__(256) void ln_gemm_wmma(
    const float* __restrict__ in0, int in0_bs,
    const float* __restrict__ in1, int in1_bs,
    const float* __restrict__ gamma, const float* __restrict__ beta,
    const h16*  __restrict__ Wh,  const float* __restrict__ bias,
    const float* __restrict__ resid, float* __restrict__ out)
{
    __shared__ float sraw[CIN * PT];  // [CIN][PT]  (32 KB)
    __shared__ h16   sxn[PT * CIN];   // [PT][CIN]  (16 KB)

    const int tid = threadIdx.x;
    const int p0  = blockIdx.x * PT;
    const int b0  = p0 / HWSZ;
    const int hw0 = p0 - b0 * HWSZ;   // PT | HWSZ, so a block never straddles b

    // hint the weight matrix into cache while the TDM streams activations
    for (int o = tid; o < O; o += 256)
        __builtin_prefetch(Wh + o * CIN, 0, 1);

    // ---- phase 1: Tensor Data Mover: 2D tile (CIN rows x PT floats) -> LDS ----
    if (tid < 32) {
        unsigned long long gaddr =
            (unsigned long long)(const void*)(in0 + (size_t)b0 * in0_bs + hw0);
        // generic pointer to __shared__ carries the LDS byte offset in low 32 bits
        unsigned int ldsoff = (unsigned int)(unsigned long long)(uintptr_t)sraw;
        u32x4 g0;
        g0.x = 1u;                                   // count=1 (valid user D#)
        g0.y = ldsoff;                               // lds_addr
        g0.z = (unsigned int)gaddr;                  // global_addr[31:0]
        g0.w = (unsigned int)(gaddr >> 32) | (2u << 30);  // addr[56:32] | type=2
        u32x8 g1;
        g1.s0 = (2u << 16);                          // data_size = 4 bytes
        g1.s1 = (unsigned int)(HWSZ & 0xFFFF) << 16; // tensor_dim0[15:0] (=0)
        g1.s2 = (unsigned int)((HWSZ >> 16) & 0xFFFF) | ((unsigned int)CIN << 16);
        g1.s3 = ((unsigned int)PT) << 16;            // tile_dim0 = PT
        g1.s4 = (unsigned int)CIN;                   // tile_dim1 = CIN
        g1.s5 = (unsigned int)HWSZ;                  // tensor_dim0_stride lo32
        g1.s6 = 0u;
        g1.s7 = 0u;
        asm volatile("tensor_load_to_lds %0, %1\n\t"
                     "s_wait_tensorcnt 0x0"
                     :: "s"(g0), "s"(g1) : "memory");
    }
    __syncthreads();

    // ---- phase 2: per-position transform -> f16 B-operand in LDS ----
    // TPP threads cooperate on one position; reductions via wave shuffles
    {
        constexpr int TPP = 256 / PT;     // 2 or 4 (power of two, <= 32)
        constexpr int CC  = CIN / TPP;    // channels per thread
        const int pp   = tid / TPP;
        const int sub  = tid % TPP;
        const int cb   = sub * CC;
        const int p    = p0 + pp;
        const int b    = p / HWSZ;
        const int hw   = p - b * HWSZ;
        if (MODE == MODE_GELU_PROD) {
            for (int i = 0; i < CC; ++i) {
                const int c = cb + i;
                float v = sraw[c * PT + pp];
                v = 0.5f * v * (1.0f + erff(v * 0.70710678118f));
                v *= in1[b * in1_bs + c * HWSZ + hw];
                sxn[pp * CIN + c] = (h16)v;
            }
        } else {
            float s1 = 0.f, s2 = 0.f;
            for (int i = 0; i < CC; ++i) {
                float v = sraw[(cb + i) * PT + pp];
                s1 += v;
                s2 += v * v;
            }
            #pragma unroll
            for (int msk = TPP >> 1; msk > 0; msk >>= 1) {
                s1 += __shfl_xor(s1, msk, 32);
                s2 += __shfl_xor(s2, msk, 32);
            }
            const float mu  = s1 * (1.0f / CIN);
            const float var = s2 * (1.0f / CIN) - mu * mu;
            const float rs  = rsqrtf(var + 1e-5f);
            for (int i = 0; i < CC; ++i) {
                const int c = cb + i;
                float v = (sraw[c * PT + pp] - mu) * rs * gamma[c] + beta[c];
                if (MODE == MODE_LN_PROD)
                    v *= in1[b * in1_bs + c * HWSZ + hw];
                sxn[pp * CIN + c] = (h16)v;
            }
        }
    }
    __syncthreads();

    // ---- phase 3: WMMA 16x16 tiles, K stepped by 32 (fully unrolled) ----
    const int wave = tid >> 5;
    const int lane = tid & 31;
    const int m    = lane & 15;   // A row M / B col N (position)
    const int half = lane >> 4;   // K-half per ISA 16-bit layout
    constexpr int oT = O >> 4, pT = PT >> 4, kC = CIN >> 5;
    constexpr int nTiles = oT * pT;

    for (int t = wave; t < nTiles; t += 8) {
        const int ot = t / pT;
        const int pt = t - ot * pT;
        v8f acc = {};
        #pragma unroll
        for (int kk = 0; kk < kC; ++kk) {
            const h16* wrow = Wh  + (ot * 16 + m) * CIN + kk * 32;
            const h16* xrow = sxn + (pt * 16 + m) * CIN + kk * 32;
            v16h A, B;
            #pragma unroll
            for (int v = 0; v < 8; ++v) {
                // ISA 16-bit A/B layout: VGPR v holds K = base, base+1
                int base = ((v < 4) ? 0 : 16) + half * 8 + (v & 3) * 2;
                v2h wa = *(const v2h*)(wrow + base);
                v2h xb = *(const v2h*)(xrow + base);
                A[2 * v]     = wa.x;  A[2 * v + 1] = wa.y;
                B[2 * v]     = xb.x;  B[2 * v + 1] = xb.y;
            }
            acc = __builtin_amdgcn_wmma_f32_16x16x32_f16(
                      false, A, false, B, (short)0, acc, false, false);
        }
        // C/D layout: lane = N + 16*(M>=8); VGPR r -> M = r + 8*(lane>=16)
        // this lane's 8 outputs are o = ot*16 + half*8 + r  (8 consecutive)
        float bb[8];
        {
            const float4* bp = (const float4*)(bias + ot * 16 + half * 8);
            ((float4*)bb)[0] = bp[0];
            ((float4*)bb)[1] = bp[1];
        }
        const int pp = pt * 16 + m;
        const int p  = p0 + pp;
        const int b  = p / HWSZ;
        const int hw = p - b * HWSZ;
        const int obase = ot * 16 + half * 8;
        #pragma unroll
        for (int r = 0; r < 8; ++r) {
            const int o = obase + r;
            float v = acc[r] + bb[r];
            if (HASR) v += resid[b * O * HWSZ + o * HWSZ + hw];
            out[b * O * HWSZ + o * HWSZ + hw] = v;
        }
    }
}

// =====================================================================
// Depthwise 3x3 conv, SAME zero padding.  in/out: [B][Cn][256][256]
// =====================================================================
__global__ __launch_bounds__(256) void dwconv3x3(
    const float* __restrict__ in, const float* __restrict__ w,
    const float* __restrict__ bias, float* __restrict__ out, int Cn)
{
    int idx = blockIdx.x * 256 + threadIdx.x;
    int totalN = NB * Cn * HWSZ;
    if (idx >= totalN) return;
    int hw = idx % HWSZ;
    int bc = idx / HWSZ;
    int c  = bc % Cn;
    int y  = hw >> 8, x = hw & 255;
    const float* base = in + (long)bc * HWSZ;
    const float* wk   = w + c * 9;
    float acc = bias[c];
    #pragma unroll
    for (int dy = -1; dy <= 1; ++dy)
        #pragma unroll
        for (int dx = -1; dx <= 1; ++dx) {
            int yy = y + dy, xx = x + dx;
            if (yy >= 0 && yy < 256 && xx >= 0 && xx < 256)
                acc += base[yy * 256 + xx] * wk[(dy + 1) * 3 + (dx + 1)];
        }
    out[idx] = acc;
}

// =====================================================================
// 8x8-patch circular convolution of q and k (== irfft2(rfft2(q)*rfft2(k)))
// 256 threads = 4 groups of 64; one patch-channel per group, LDS staged.
// =====================================================================
__global__ __launch_bounds__(256) void patch_circ_corr(
    const float* __restrict__ q, int q_bs,
    const float* __restrict__ k, int k_bs,
    float* __restrict__ out, int out_bs, int Cn)
{
    __shared__ float sq[4][64];
    __shared__ float sk[4][64];
    const int g = threadIdx.x >> 6, t = threadIdx.x & 63;
    const int pc = blockIdx.x * 4 + g;            // < NB*Cn*1024
    const int patch = pc & 1023;
    const int bc    = pc >> 10;
    const int c  = bc % Cn, b = bc / Cn;
    const int py = patch >> 5, px = patch & 31;
    const int my = t >> 3, mx = t & 7;
    const int pix = (py * 8 + my) * 256 + px * 8 + mx;
    sq[g][t] = q[b * q_bs + c * HWSZ + pix];
    sk[g][t] = k[b * k_bs + c * HWSZ + pix];
    __syncthreads();
    float acc = 0.f;
    #pragma unroll
    for (int i = 0; i < 8; ++i)
        #pragma unroll
        for (int j = 0; j < 8; ++j)
            acc += sq[g][i * 8 + j] * sk[g][((my - i) & 7) * 8 + ((mx - j) & 7)];
    out[b * out_bs + c * HWSZ + pix] = acc;
}

// =====================================================================
// 8x8-patch circular convolution with per-channel spatial kernel gk[c][8][8]
// (== irfft2(rfft2(y) * F) with real filter F)
// =====================================================================
__global__ __launch_bounds__(256) void patch_circ_filter(
    const float* __restrict__ in, int in_bs,
    const float* __restrict__ gk,
    float* __restrict__ out, int out_bs, int Cn)
{
    __shared__ float sq[4][64];
    __shared__ float sg[4][64];
    const int g = threadIdx.x >> 6, t = threadIdx.x & 63;
    const int pc = blockIdx.x * 4 + g;
    const int patch = pc & 1023;
    const int bc    = pc >> 10;
    const int c  = bc % Cn, b = bc / Cn;
    const int py = patch >> 5, px = patch & 31;
    const int my = t >> 3, mx = t & 7;
    const int pix = (py * 8 + my) * 256 + px * 8 + mx;
    sq[g][t] = in[b * in_bs + c * HWSZ + pix];
    sg[g][t] = gk[c * 64 + t];
    __syncthreads();
    float acc = 0.f;
    #pragma unroll
    for (int i = 0; i < 8; ++i)
        #pragma unroll
        for (int j = 0; j < 8; ++j)
            acc += sq[g][i * 8 + j] * sg[g][((my - i) & 7) * 8 + ((mx - j) & 7)];
    out[b * out_bs + c * HWSZ + pix] = acc;
}

// =====================================================================
// Build per-channel 8x8 spatial kernel from real rfft2 filter F[c][8][5]
// using the Hermitian extension implied by irfft2.
// =====================================================================
__global__ __launch_bounds__(256) void build_spatial_filter(
    const float* __restrict__ F, float* __restrict__ gk, int Cn)
{
    int idx = blockIdx.x * 256 + threadIdx.x;
    if (idx >= Cn * 64) return;
    int c  = idx >> 6;
    int mn = idx & 63;
    int mm = mn >> 3, nn = mn & 7;
    const float ct[8] = {1.f, 0.70710678118f, 0.f, -0.70710678118f,
                         -1.f, -0.70710678118f, 0.f, 0.70710678118f};
    float acc = 0.f;
    for (int u = 0; u < 8; ++u)
        for (int v = 0; v < 8; ++v) {
            float f = (v <= 4) ? F[c * 40 + u * 5 + v]
                               : F[c * 40 + ((8 - u) & 7) * 5 + (8 - v)];
            acc += f * ct[(u * mm + v * nn) & 7];
        }
    gk[idx] = acc * (1.0f / 64.0f);
}

// =====================================================================
// f32 -> f16 weight conversion
// =====================================================================
__global__ __launch_bounds__(256) void f32_to_f16(
    const float* __restrict__ in, h16* __restrict__ out, int n)
{
    int i = blockIdx.x * 256 + threadIdx.x;
    if (i < n) out[i] = (h16)in[i];
}

// =====================================================================
extern "C" void kernel_launch(void* const* d_in, const int* in_sizes, int n_in,
                              void* d_out, int out_size, void* d_ws, size_t ws_size,
                              hipStream_t stream)
{
    (void)in_sizes; (void)n_in; (void)out_size; (void)ws_size;

    const float* x        = (const float*)d_in[0];
    const float* ln1_g    = (const float*)d_in[1];
    const float* ln1_b    = (const float*)d_in[2];
    const float* att_hid_w= (const float*)d_in[3];
    const float* att_hid_b= (const float*)d_in[4];
    const float* att_dw_w = (const float*)d_in[5];
    const float* att_dw_b = (const float*)d_in[6];
    const float* att_n_g  = (const float*)d_in[7];
    const float* att_n_b  = (const float*)d_in[8];
    const float* att_out_w= (const float*)d_in[9];
    const float* att_out_b= (const float*)d_in[10];
    const float* ln2_g    = (const float*)d_in[11];
    const float* ln2_b    = (const float*)d_in[12];
    const float* ffn_in_w = (const float*)d_in[13];
    const float* ffn_in_b = (const float*)d_in[14];
    const float* ffn_fft  = (const float*)d_in[15];
    const float* ffn_dw_w = (const float*)d_in[16];
    const float* ffn_dw_b = (const float*)d_in[17];
    const float* ffn_out_w= (const float*)d_in[18];
    const float* ffn_out_b= (const float*)d_in[19];
    float* out = (float*)d_out;

    // ---- workspace layout ----
    const size_t szc = (size_t)NB * HWSZ;      // elems per channel over batch
    float* R0 = (float*)d_ws;                  // 384 ch: h -> corr -> y -> z
    float* R1 = R0 + 384 * szc;                // 384 ch: qkv -> y2
    float* R2 = R1 + 384 * szc;                // 64  ch: x2 (attention output)
    float* gk = R2 + 64 * szc;                 // 256*64 spatial filter
    h16* wh_hid = (h16*)(gk + 256 * 64);       // 384*64
    h16* wh_att = wh_hid + 384 * 64;           // 64*128
    h16* wh_fin = wh_att + 64 * 128;           // 256*64
    h16* wh_fout= wh_fin + 256 * 64;           // 64*128

    // ---- 0) weight conversion + spectral filter prep ----
    f32_to_f16<<<(384*64 + 255)/256, 256, 0, stream>>>(att_hid_w, wh_hid, 384*64);
    f32_to_f16<<<(64*128 + 255)/256, 256, 0, stream>>>(att_out_w, wh_att, 64*128);
    f32_to_f16<<<(256*64 + 255)/256, 256, 0, stream>>>(ffn_in_w, wh_fin, 256*64);
    f32_to_f16<<<(64*128 + 255)/256, 256, 0, stream>>>(ffn_out_w, wh_fout, 64*128);
    build_spatial_filter<<<(256*64 + 255)/256, 256, 0, stream>>>(ffn_fft, gk, 256);

    // ---- 1) h = conv1x1(LN(x)) : [B,384,H,W] ----
    ln_gemm_wmma<MODE_LN, 64, 384, 128, false><<<NPOS/128, 256, 0, stream>>>(
        x, 64*HWSZ, nullptr, 0, ln1_g, ln1_b, wh_hid, att_hid_b, nullptr, R0);

    // ---- 2) qkv = dwconv3(h) ----
    dwconv3x3<<<(NB*384*HWSZ)/256, 256, 0, stream>>>(R0, att_dw_w, att_dw_b, R1, 384);

    // ---- 3) corr = per-patch circular conv(q,k) : [B,128,H,W] ----
    patch_circ_corr<<<(NB*128*1024)/4, 256, 0, stream>>>(
        R1, 384*HWSZ,                 // q = qkv[:, 0:128]
        R1 + 128*HWSZ, 384*HWSZ,      // k = qkv[:, 128:256]
        R0, 128*HWSZ, 128);

    // ---- 4) x2 = x + conv1x1(v * LN(corr)) : [B,64,H,W] ----
    ln_gemm_wmma<MODE_LN_PROD, 128, 64, 64, true><<<NPOS/64, 256, 0, stream>>>(
        R0, 128*HWSZ, R1 + 256*HWSZ, 384*HWSZ,  // in1 = v = qkv[:, 256:384]
        att_n_g, att_n_b, wh_att, att_out_b, x, R2);

    // ---- 5) y = conv1x1(LN(x2)) : [B,256,H,W] ----
    ln_gemm_wmma<MODE_LN, 64, 256, 128, false><<<NPOS/128, 256, 0, stream>>>(
        R2, 64*HWSZ, nullptr, 0, ln2_g, ln2_b, wh_fin, ffn_in_b, nullptr, R0);

    // ---- 6) y2 = per-patch circular conv with learned spectral filter ----
    patch_circ_filter<<<(NB*256*1024)/4, 256, 0, stream>>>(
        R0, 256*HWSZ, gk, R1, 256*HWSZ, 256);

    // ---- 7) z = dwconv3(y2) : [B,256,H,W] ----
    dwconv3x3<<<(NB*256*HWSZ)/256, 256, 0, stream>>>(R1, ffn_dw_w, ffn_dw_b, R0, 256);

    // ---- 8) out = x2 + conv1x1(gelu(z1) * z2) : [B,64,H,W] ----
    ln_gemm_wmma<MODE_GELU_PROD, 128, 64, 64, true><<<NPOS/64, 256, 0, stream>>>(
        R0, 256*HWSZ, R0 + 128*HWSZ, 256*HWSZ,  // z1, z2
        nullptr, nullptr, wh_fout, ffn_out_b, R2, out);
}
